// VectorQuantizer_86449101734161
// MI455X (gfx1250) — compile-verified, matrix operand order checked by reading, not run on hardware
//
#include <hip/hip_runtime.h>

// ---- CDNA5 WMMA vector types ----
typedef __attribute__((ext_vector_type(16))) __bf16       v16bf;
typedef __attribute__((ext_vector_type(8)))  float        v8f;
typedef __attribute__((ext_vector_type(8)))  unsigned int v8u;

#define NUM_EMB  512
#define EMB_DIM  64
#define HW       4096                    // H*W
#define BSTRIDE  (EMB_DIM * HW)          // per-batch stride in z_e / out
#define NROWS    131072                  // 32*64*64 flat vectors
#define NTILES   (NROWS / 16)            // 8192 row tiles -> 4096 pairs
#define ROW_U    36                      // dwords per packed-bf16 codebook row (144B, bank-optimal)
#define LDS_LO   (NUM_EMB * ROW_U)       // lo-plane offset (dwords)
#define LDS_C2   (2 * NUM_EMB * ROW_U)   // c2 offset (dwords)
#define LDS_U32  (LDS_C2 + NUM_EMB)      // 37376 dwords = 149504 B

static __device__ __forceinline__ unsigned int f2u(float x) { return __builtin_bit_cast(unsigned int, x); }
static __device__ __forceinline__ float        u2f(unsigned int x) { return __builtin_bit_cast(float, x); }

// split (x0,x1) into packed-bf16 hi plane (truncated top 16 bits) + bf16 residual plane
static __device__ __forceinline__ void split2(float x0, float x1,
                                              unsigned int& hi_pk, unsigned int& lo_pk) {
    unsigned int h0 = f2u(x0) & 0xFFFF0000u;
    unsigned int h1 = f2u(x1) & 0xFFFF0000u;
    float l0 = x0 - u2f(h0);                       // exact in f32
    float l1 = x1 - u2f(h1);
    hi_pk = (h0 >> 16) | h1;
    lo_pk = ((f2u(l0) & 0xFFFF0000u) >> 16) | (f2u(l1) & 0xFFFF0000u);
}

static __device__ __forceinline__ float bflo(unsigned int u) { return u2f(u << 16); }         // low bf16 -> f32
static __device__ __forceinline__ float bfhi(unsigned int u) { return u2f(u & 0xFFFF0000u); } // high bf16 -> f32

// d-index owned by this lane for (chunk c, pair p); pair covers d0, d0+1 (A/B 16-bit WMMA layout)
static __device__ __forceinline__ int dpair(int c, int p, int g) {
    return 32 * c + ((p < 4) ? (8 * g + 2 * p) : (16 + 8 * g + 2 * (p - 4)));
}

__global__ __launch_bounds__(256)
__attribute__((amdgpu_waves_per_eu(4)))   // cap VGPRs <= file/4: stay under 256, no s_set_vgpr_msb
void vq_main(const float* __restrict__ z_e,
             float* __restrict__ out,
             float* __restrict__ loss_acc,
             float* __restrict__ counts,
             const float* __restrict__ cb)
{
    extern __shared__ unsigned int ldsu[];
    float* c2s = (float*)&ldsu[LDS_C2];
    const int tid = threadIdx.x;

    // ---- Stage codebook into LDS as packed bf16 hi/lo planes; exact f32 norms ----
    for (int k = tid; k < NUM_EMB; k += 256) {
        const float* row = cb + k * EMB_DIM;
        float s = 0.f;
#pragma unroll 8
        for (int j = 0; j < EMB_DIM / 2; ++j) {
            float x0 = row[2 * j], x1 = row[2 * j + 1];
            s += x0 * x0 + x1 * x1;
            unsigned int hp, lp;
            split2(x0, x1, hp, lp);
            ldsu[k * ROW_U + j]          = hp;
            ldsu[LDS_LO + k * ROW_U + j] = lp;
        }
        c2s[k] = s;
    }
    __syncthreads();

    const int lane = tid & 31;
    const int wave = tid >> 5;
    const int g    = lane >> 4;   // K sub-group half
    const int m    = lane & 15;   // row in A tile / column in B tile

    // one M=32 super-tile (two 16-row tiles sharing B fragments) per wave
    const int pair = blockIdx.x * 8 + wave;          // 0..4095

    // ---- Load + split A fragments for both tiles (scalar coalesced loads, d-stride = HW) ----
    v8u ahi_u[2][2], alo_u[2][2];                    // [tile u][chunk c]
#pragma unroll
    for (int u = 0; u < 2; ++u) {
        const int n = pair * 32 + u * 16 + m;
        const float* zb = z_e + (long)(n >> 12) * BSTRIDE + (n & (HW - 1));
#pragma unroll
        for (int c = 0; c < 2; ++c) {
#pragma unroll
            for (int p = 0; p < 8; ++p) {
                const int d0 = dpair(c, p, g);
                const float x0 = __builtin_nontemporal_load(zb + (long)d0 * HW);
                const float x1 = __builtin_nontemporal_load(zb + (long)(d0 + 1) * HW);
                unsigned int hp, lp;
                split2(x0, x1, hp, lp);
                ahi_u[u][c][p] = hp;
                alo_u[u][c][p] = lp;
            }
        }
    }

    float minv[2][8];
    int   mini[2][8];
#pragma unroll
    for (int u = 0; u < 2; ++u)
#pragma unroll
        for (int r = 0; r < 8; ++r) { minv[u][r] = 3.4e38f; mini[u][r] = 0; }

    // ---- 32 code tiles; split-bf16 GEMM: 12 chained v_wmma_f32_16x16x32_bf16 per tile pair ----
    for (int t = 0; t < NUM_EMB / 16; ++t) {
        const int code = t * 16 + m;
        v8f acc0 = {0.f, 0.f, 0.f, 0.f, 0.f, 0.f, 0.f, 0.f};
        v8f acc1 = {0.f, 0.f, 0.f, 0.f, 0.f, 0.f, 0.f, 0.f};
#pragma unroll
        for (int c = 0; c < 2; ++c) {
            const unsigned int* bp = &ldsu[code * ROW_U + 16 * c + 4 * g];
            const uint4 h0 = *(const uint4*)(bp);                 // hi plane, VGPR0-3
            const uint4 h1 = *(const uint4*)(bp + 8);             // hi plane, VGPR4-7
            const uint4 l0 = *(const uint4*)(bp + LDS_LO);        // lo plane, VGPR0-3
            const uint4 l1 = *(const uint4*)(bp + LDS_LO + 8);    // lo plane, VGPR4-7
            v8u bhu = {h0.x, h0.y, h0.z, h0.w, h1.x, h1.y, h1.z, h1.w};
            v8u blu = {l0.x, l0.y, l0.z, l0.w, l1.x, l1.y, l1.z, l1.w};
            v16bf bhi = __builtin_bit_cast(v16bf, bhu);
            v16bf blo = __builtin_bit_cast(v16bf, blu);
            v16bf a0h = __builtin_bit_cast(v16bf, ahi_u[0][c]);
            v16bf a0l = __builtin_bit_cast(v16bf, alo_u[0][c]);
            v16bf a1h = __builtin_bit_cast(v16bf, ahi_u[1][c]);
            v16bf a1l = __builtin_bit_cast(v16bf, alo_u[1][c]);
            // two independent accumulator chains share each B fragment
            acc0 = __builtin_amdgcn_wmma_f32_16x16x32_bf16(false, a0h, false, bhi, (short)0, acc0, false, false);
            acc1 = __builtin_amdgcn_wmma_f32_16x16x32_bf16(false, a1h, false, bhi, (short)0, acc1, false, false);
            acc0 = __builtin_amdgcn_wmma_f32_16x16x32_bf16(false, a0h, false, blo, (short)0, acc0, false, false);
            acc1 = __builtin_amdgcn_wmma_f32_16x16x32_bf16(false, a1h, false, blo, (short)0, acc1, false, false);
            acc0 = __builtin_amdgcn_wmma_f32_16x16x32_bf16(false, a0l, false, bhi, (short)0, acc0, false, false);
            acc1 = __builtin_amdgcn_wmma_f32_16x16x32_bf16(false, a1l, false, bhi, (short)0, acc1, false, false);
        }
        const float c2 = c2s[code];
#pragma unroll
        for (int r = 0; r < 8; ++r) {
            const float s0 = c2 - 2.0f * acc0[r];
            if (s0 < minv[0][r]) { minv[0][r] = s0; mini[0][r] = code; }
            const float s1 = c2 - 2.0f * acc1[r];
            if (s1 < minv[1][r]) { minv[1][r] = s1; mini[1][r] = code; }
        }
    }

    // ---- per-tile argmin, broadcast, histogram, gather/output/loss ----
    float err = 0.f;
#pragma unroll
    for (int u = 0; u < 2; ++u) {
#pragma unroll
        for (int r = 0; r < 8; ++r) {
            float v = minv[u][r]; int i = mini[u][r];
#pragma unroll
            for (int off = 8; off >= 1; off >>= 1) {
                const float ov = __shfl_xor(v, off, 32);
                const int   oi = __shfl_xor(i, off, 32);
                if (ov < v || (ov == v && oi < i)) { v = ov; i = oi; }
            }
            mini[u][r] = i;
        }
        int myidx = 0;
#pragma unroll
        for (int r = 0; r < 8; ++r) {
            const int lo = __shfl(mini[u][r], 0, 32);   // row r
            const int hi = __shfl(mini[u][r], 16, 32);  // row r+8
            if (m == r)     myidx = lo;
            if (m == r + 8) myidx = hi;
        }
        if (g == 0) atomicAdd(&counts[myidx], 1.0f);

        const int n = pair * 32 + u * 16 + m;
        float* ob = out + (long)(n >> 12) * BSTRIDE + (n & (HW - 1));
        const unsigned int* qh = &ldsu[myidx * ROW_U];
#pragma unroll
        for (int c = 0; c < 2; ++c) {
#pragma unroll
            for (int p = 0; p < 8; ++p) {
                const int d0 = dpair(c, p, g);
                const unsigned int uh = qh[d0 >> 1];
                const unsigned int ul = qh[LDS_LO + (d0 >> 1)];
                const float q0 = bflo(uh) + bflo(ul);
                const float q1 = bfhi(uh) + bfhi(ul);
                // reconstruct z from its hi/lo split (error ~2^-17 relative)
                const float z0 = bflo(ahi_u[u][c][p]) + bflo(alo_u[u][c][p]);
                const float z1 = bfhi(ahi_u[u][c][p]) + bfhi(alo_u[u][c][p]);
                const float e0 = q0 - z0;
                const float e1 = q1 - z1;
                err += e0 * e0 + e1 * e1;
                __builtin_nontemporal_store(q0, ob + (long)d0 * HW);
                __builtin_nontemporal_store(q1, ob + (long)(d0 + 1) * HW);
            }
        }
    }
#pragma unroll
    for (int off = 16; off >= 1; off >>= 1) err += __shfl_xor(err, off, 32);
    if (lane == 0) atomicAdd(loss_acc, err);
}

// Finalize: vq_loss, perplexity, num_restarts into the output tail.
__global__ __launch_bounds__(512) void vq_finalize(const float* __restrict__ loss_acc,
                                                   const float* __restrict__ counts,
                                                   float* __restrict__ out_tail)
{
    __shared__ float red[512];
    const int k = threadIdx.x;
    const float p = counts[k] * (1.0f / (float)NROWS);
    red[k] = p * logf(p + 1e-10f);
    __syncthreads();
    for (int s = 256; s >= 1; s >>= 1) {
        if (k < s) red[k] += red[k + s];
        __syncthreads();
    }
    if (k == 0) {
        out_tail[0] = loss_acc[0] * 1.25f / (float)((long)NROWS * EMB_DIM); // q + 0.25*e latent loss
        out_tail[1] = expf(-red[0]);                                        // perplexity
        out_tail[2] = 0.0f;                                                 // num_restarts
    }
}

extern "C" void kernel_launch(void* const* d_in, const int* in_sizes, int n_in,
                              void* d_out, int out_size, void* d_ws, size_t ws_size,
                              hipStream_t stream) {
    const float* z_e = (const float*)d_in[0];
    const float* cb  = (const float*)d_in[1];
    float* out = (float*)d_out;
    float* ws  = (float*)d_ws;   // [0]=loss accumulator, [1..512]=counts

    hipMemsetAsync(ws, 0, (1 + NUM_EMB) * sizeof(float), stream);

    const int blocks   = NTILES / 16;               // 512 blocks; 8 waves x 1 pair (M=32) each
    const size_t shmem = (size_t)LDS_U32 * 4;       // 149504 B dynamic LDS

    vq_main<<<blocks, 256, shmem, stream>>>(z_e, out, ws, ws + 1, cb);
    vq_finalize<<<1, 512, 0, stream>>>(ws, ws + 1, out + (size_t)NROWS * EMB_DIM);
}
